// CrossAttention_24163486007414
// MI455X (gfx1250) — matrix operand order repaired and checked by reading, not compiled, not run on hardware
//
#include <hip/hip_runtime.h>
#include <hip/hip_bf16.h>
#include <stdint.h>

#define HEADS 8
#define DIM_HEAD 64
#define TOPK 32
#define NTOK 4096
#define DMODEL 512
#define BATCH 2
#define SCALE 0.125f
#define MROWS (BATCH * NTOK) /* 8192 */

typedef __attribute__((ext_vector_type(16))) __bf16 v16bf;
typedef __attribute__((ext_vector_type(8)))  float  v8f;
typedef __attribute__((ext_vector_type(8)))  unsigned short v8us;
typedef __attribute__((ext_vector_type(16))) unsigned short v16us;

static __device__ __forceinline__ unsigned short f2bf(float f) {
  unsigned int u = __float_as_uint(f);
  u += 0x7fffu + ((u >> 16) & 1u);          // round-to-nearest-even
  return (unsigned short)(u >> 16);
}

static __device__ __forceinline__ v16bf frag_cast(v8us lo, v8us hi) {
  v16us full = __builtin_shufflevector(lo, hi, 0,1,2,3,4,5,6,7,8,9,10,11,12,13,14,15);
  union { v16us u; v16bf b; } cv; cv.u = full; return cv.b;
}

// A fragment 16x32 bf16: lane (l&15)=row M; K runs [(l>>4)*8,+8) and [16+(l>>4)*8,+8)
static __device__ __forceinline__ v16bf load_fragA(const unsigned short* rowptr, int kk, int lane) {
  const unsigned short* p = rowptr + kk + ((lane >> 4) << 3);
  v8us lo = *(const v8us*)p;
  v8us hi = *(const v8us*)(p + 16);
  return frag_cast(lo, hi);
}

// B fragment 32x16 bf16: lane (l&15)=column N; K runs [(l>>4)*8,+8) and [16+(l>>4)*8,+8)
static __device__ __forceinline__ v16bf load_fragB(const unsigned short* colptr, int kk, int lane) {
  const unsigned short* p = colptr + kk + ((lane >> 4) << 3);
  v8us lo = *(const v8us*)p;
  v8us hi = *(const v8us*)(p + 16);
  return frag_cast(lo, hi);
}

static __device__ __forceinline__ v8f wmma_bf16(v16bf a, v16bf b, v8f c) {
  return __builtin_amdgcn_wmma_f32_16x16x32_bf16(false, a, false, b, (short)0, c, false, false);
}

// ---------------- conversion kernels ----------------
__global__ __launch_bounds__(256)
void k_f32_to_bf16(const float* __restrict__ src, unsigned short* __restrict__ dst, int n) {
  int i = blockIdx.x * 256 + threadIdx.x;
  if (i < n) dst[i] = f2bf(src[i]);
}

__global__ __launch_bounds__(256)
void k_transpose_bf16(const float* __restrict__ src, unsigned short* __restrict__ dst) {
  // src: W[k][n] (512x512); dst: Wt[n][k] bf16, so WMMA B-columns are contiguous rows
  int i = blockIdx.x * 256 + threadIdx.x;
  int n = i >> 9, k = i & 511;
  dst[i] = f2bf(src[(size_t)k * DMODEL + n]);
}

// ---------------- generic 64x64-tile bf16 WMMA GEMM ----------------
// C[M x 512] = A[M x 512](bf16 rowmajor) * Bt^T, Bt[n][k] bf16.
// mode 0: outF[row*512+col] = v + bias[col]
// mode 1: outH at head layout [b,h,n,dh] (bf16)   (Q, K)
// mode 2: outF at head layout [b,h,n,dh] (f32)    (V)
__global__ __launch_bounds__(128)
void k_gemm64(const unsigned short* __restrict__ A, const unsigned short* __restrict__ Bt,
              float* __restrict__ outF, unsigned short* __restrict__ outH,
              const float* __restrict__ bias, int mode) {
  const int lane = threadIdx.x & 31;
  const int w    = threadIdx.x >> 5;
  const int rowbase = blockIdx.x * 64 + w * 16;
  const int colbase = blockIdx.y * 64;
  const unsigned short* Arow = A + (size_t)(rowbase + (lane & 15)) * DMODEL;

  v8f c0 = {}, c1 = {}, c2 = {}, c3 = {};
  for (int kk = 0; kk < DMODEL; kk += 32) {
    v16bf a  = load_fragA(Arow, kk, lane);
    v16bf b0 = load_fragB(Bt + (size_t)(colbase +  0 + (lane & 15)) * DMODEL, kk, lane);
    v16bf b1 = load_fragB(Bt + (size_t)(colbase + 16 + (lane & 15)) * DMODEL, kk, lane);
    v16bf b2 = load_fragB(Bt + (size_t)(colbase + 32 + (lane & 15)) * DMODEL, kk, lane);
    v16bf b3 = load_fragB(Bt + (size_t)(colbase + 48 + (lane & 15)) * DMODEL, kk, lane);
    c0 = wmma_bf16(a, b0, c0);
    c1 = wmma_bf16(a, b1, c1);
    c2 = wmma_bf16(a, b2, c2);
    c3 = wmma_bf16(a, b3, c3);
  }
  v8f cs[4] = {c0, c1, c2, c3};
  const int mo = (lane >> 4) << 3;
  for (int j = 0; j < 4; ++j) {
    #pragma unroll
    for (int r = 0; r < 8; ++r) {
      int row = rowbase + r + mo;
      int col = colbase + j * 16 + (lane & 15);
      float v = cs[j][r];
      if (mode == 0) {
        outF[(size_t)row * DMODEL + col] = v + bias[col];
      } else {
        int bi = row >> 12, ni = row & (NTOK - 1);
        int hi = col >> 6,  dhi = col & 63;
        size_t di = (((size_t)(bi * HEADS + hi) * NTOK) + ni) * DIM_HEAD + dhi;
        if (mode == 1) outH[di] = f2bf(v);
        else           outF[di] = v;
      }
    }
  }
}

// ---------------- streaming top-k attention ----------------
// grid: (NTOK/64, HEADS, BATCH), block 128 (4 waves). Each WG: 64 queries.
__global__ __launch_bounds__(128)
void k_attn(const unsigned short* __restrict__ Qb, const unsigned short* __restrict__ Kb,
            const float* __restrict__ Vf, unsigned short* __restrict__ Ob) {
  __shared__ __align__(16) unsigned short Ks[128 * 64]; // key chunk, 16KB
  __shared__ __align__(16) float sc[64 * 128];          // score tile, 32KB
  __shared__ float lv[64 * TOPK];                       // top-k values, 8KB
  __shared__ int   li[64 * TOPK];                       // top-k indices, 8KB

  const int t = threadIdx.x;
  const int lane = t & 31;
  const int w = t >> 5;
  const int qblk = blockIdx.x, hd = blockIdx.y, bi = blockIdx.z;
  const size_t qkbase = (size_t)(bi * HEADS + hd) * NTOK;
  const int qbase = qblk * 64;

  // this wave's Q strip fragments (dh=64 -> two k-steps), kept in registers
  const unsigned short* Qrow = Qb + (qkbase + qbase + w * 16 + (lane & 15)) * DIM_HEAD;
  v16bf a0 = load_fragA(Qrow, 0, lane);
  v16bf a1 = load_fragA(Qrow, 32, lane);

  for (int i = t; i < 64 * TOPK; i += 128) { lv[i] = -3.0e38f; li[i] = 0; }
  float minv = -3.0e38f; int minpos = 0;   // private to the query-owning thread (t<64)

  for (int ch = 0; ch < NTOK / 128; ++ch) {
    const int kbase = ch * 128;
    __syncthreads();
    { // stage 128 key rows into LDS (one row per thread, 128B each)
      const unsigned short* src = Kb + (qkbase + kbase + t) * DIM_HEAD;
      unsigned short* dst = &Ks[t * 64];
      #pragma unroll
      for (int i = 0; i < 8; ++i) ((v8us*)dst)[i] = ((const v8us*)src)[i];
    }
    __syncthreads();
    // scores: wave w computes rows [w*16,+16) x 128 cols: 8 tiles x 2 WMMAs
    #pragma unroll
    for (int j = 0; j < 8; ++j) {
      const unsigned short* colp = &Ks[(j * 16 + (lane & 15)) * 64];
      v16bf b0 = load_fragB(colp, 0, lane);
      v16bf b1 = load_fragB(colp, 32, lane);
      v8f c = {};
      c = wmma_bf16(a0, b0, c);
      c = wmma_bf16(a1, b1, c);
      const int mo = (lane >> 4) << 3;
      #pragma unroll
      for (int r = 0; r < 8; ++r)
        sc[(w * 16 + r + mo) * 128 + j * 16 + (lane & 15)] = c[r] * SCALE;
    }
    __syncthreads();
    // running top-32 update: one thread per query
    if (t < 64) {
      float* qv = &lv[t * TOPK];
      int*   qi = &li[t * TOPK];
      const float* row = &sc[t * 128];
      for (int i = 0; i < 128; ++i) {
        float v = row[i];
        if (v > minv) {
          qv[minpos] = v; qi[minpos] = kbase + i;
          float m = qv[0]; int mp = 0;
          #pragma unroll
          for (int j2 = 1; j2 < TOPK; ++j2) { float x = qv[j2]; if (x < m) { m = x; mp = j2; } }
          minv = m; minpos = mp;
        }
      }
    }
  }
  __syncthreads();
  // softmax over the 32 kept scores (in place: lv becomes weights)
  if (t < 64) {
    float* qv = &lv[t * TOPK];
    float mx = -3.0e38f;
    #pragma unroll
    for (int j = 0; j < TOPK; ++j) mx = fmaxf(mx, qv[j]);
    float s = 0.f;
    #pragma unroll
    for (int j = 0; j < TOPK; ++j) { float e = __expf(qv[j] - mx); qv[j] = e; s += e; }
    float inv = 1.f / s;
    #pragma unroll
    for (int j = 0; j < TOPK; ++j) qv[j] *= inv;
  }
  __syncthreads();
  // sparse weighted gather of V: 2 threads per query, 32 dh lanes each
  {
    const int q = t >> 1;
    const int half = (t & 1) * 32;
    const float* Vbase = Vf + qkbase * DIM_HEAD;
    float acc[32];
    #pragma unroll
    for (int d = 0; d < 32; ++d) acc[d] = 0.f;
    for (int j = 0; j < TOPK; ++j) {
      float wgt = lv[q * TOPK + j];
      const float* vr = Vbase + (size_t)li[q * TOPK + j] * DIM_HEAD + half;
      #pragma unroll
      for (int d = 0; d < 32; ++d) acc[d] += wgt * vr[d];
    }
    unsigned short* orow = Ob + (size_t)(bi * NTOK + qbase + q) * DMODEL + hd * DIM_HEAD + half;
    #pragma unroll
    for (int d = 0; d < 32; ++d) orow[d] = f2bf(acc[d]);
  }
}

extern "C" void kernel_launch(void* const* d_in, const int* in_sizes, int n_in,
                              void* d_out, int out_size, void* d_ws, size_t ws_size,
                              hipStream_t stream) {
  (void)in_sizes; (void)n_in; (void)out_size; (void)ws_size;
  const float* x  = (const float*)d_in[0];
  const float* Wq = (const float*)d_in[1];
  const float* Wk = (const float*)d_in[2];
  const float* Wv = (const float*)d_in[3];
  const float* Wo = (const float*)d_in[4];
  const float* bo = (const float*)d_in[5];
  float* out = (float*)d_out;

  char* ws = (char*)d_ws;
  size_t off = 0;
  auto alloc = [&](size_t bytes) -> char* {
    char* p = ws + off; off += (bytes + 255) & ~(size_t)255; return p;
  };
  unsigned short* xb  = (unsigned short*)alloc((size_t)MROWS * DMODEL * 2);
  unsigned short* Wqt = (unsigned short*)alloc((size_t)DMODEL * DMODEL * 2);
  unsigned short* Wkt = (unsigned short*)alloc((size_t)DMODEL * DMODEL * 2);
  unsigned short* Wvt = (unsigned short*)alloc((size_t)DMODEL * DMODEL * 2);
  unsigned short* Wot = (unsigned short*)alloc((size_t)DMODEL * DMODEL * 2);
  unsigned short* Qbf = (unsigned short*)alloc((size_t)MROWS * DMODEL * 2);
  unsigned short* Kbf = (unsigned short*)alloc((size_t)MROWS * DMODEL * 2);
  float*          Vff = (float*)alloc((size_t)MROWS * DMODEL * 4);
  unsigned short* Obf = (unsigned short*)alloc((size_t)MROWS * DMODEL * 2);

  k_f32_to_bf16<<<(MROWS * DMODEL) / 256, 256, 0, stream>>>(x, xb, MROWS * DMODEL);
  k_transpose_bf16<<<(DMODEL * DMODEL) / 256, 256, 0, stream>>>(Wq, Wqt);
  k_transpose_bf16<<<(DMODEL * DMODEL) / 256, 256, 0, stream>>>(Wk, Wkt);
  k_transpose_bf16<<<(DMODEL * DMODEL) / 256, 256, 0, stream>>>(Wv, Wvt);
  k_transpose_bf16<<<(DMODEL * DMODEL) / 256, 256, 0, stream>>>(Wo, Wot);

  dim3 gg(MROWS / 64, DMODEL / 64);
  k_gemm64<<<gg, 128, 0, stream>>>(xb, Wqt, nullptr, Qbf, nullptr, 1);
  k_gemm64<<<gg, 128, 0, stream>>>(xb, Wkt, nullptr, Kbf, nullptr, 1);
  k_gemm64<<<gg, 128, 0, stream>>>(xb, Wvt, Vff, nullptr, nullptr, 2);

  k_attn<<<dim3(NTOK / 64, HEADS, BATCH), 128, 0, stream>>>(Qbf, Kbf, Vff, Obf);

  k_gemm64<<<gg, 128, 0, stream>>>(Obf, Wot, out, nullptr, bo, 0);
}
